// ContextPointEncoder_21414706938285
// MI455X (gfx1250) — matrix operand order repaired and checked by reading, not compiled, not run on hardware
//
#include <hip/hip_runtime.h>

// ---------------------------------------------------------------------------
// ContextPointEncoder fused pipeline for MI455X (gfx1250, wave32, WMMA bf16)
//
// B=128, N_OBJ=128, HIDDEN=128, N_FEAT=4.  Heavy work: two 2M x 128 x 128
// GEMMs. Everything is fused in LDS; only per-channel BN statistics and the
// small (16384x128) tensors a / s_pre / v_rel / x4 touch HBM (~26 MB ws).
// ---------------------------------------------------------------------------

#define NB    128
#define NO    128
#define HD    128
#define NF    4
#define NROW  (NB*NO)              // 16384
#define NPAIRF 2097152.0f          // B * n * n
#define EPSBN 1e-5f
#define NWG_PAIR 1024              // workgroups for the heavy pair passes

typedef __bf16 bf16_t;
typedef __attribute__((ext_vector_type(16))) __bf16 v16bf;
typedef __attribute__((ext_vector_type(8)))  __bf16 v8bf;
typedef __attribute__((ext_vector_type(8)))  float  v8f;

// ---------------- workspace layout (bytes), total ~26.5 MiB ----------------
#define WS_A      0                              // a = pts@W1^T (f32, 8MiB); reused for x4
#define WS_S      (WS_A    + NROW*HD*4)          // s_pre = pts@W0^T (8MiB)
#define WS_VREL   (WS_S    + NROW*HD*4)          // v_rel (8MiB)
#define WS_W2BF   (WS_VREL + NROW*HD*4)          // bf16 weights, 32KB each
#define WS_W3BF   (WS_W2BF + HD*HD*2)
#define WS_W4BF   (WS_W3BF + HD*HD*2)
#define WS_VAR1P  (WS_W4BF + HD*HD*2)            // [NB][HD] layer-1 var partials
#define WS_SSUMP  (WS_VAR1P + NB*HD*4)           // [NB][HD] s-path sum partials
#define WS_SSQP   (WS_SSUMP + NB*HD*4)           // [NB][HD] s-path sumsq partials
#define WS_PART2  (WS_SSQP  + NB*HD*4)           // [NWG_PAIR][256] x2 stats partials
#define WS_PART3  (WS_PART2 + NWG_PAIR*256*4)    // [NWG_PAIR][256] x3 stats partials
#define WS_PART4  (WS_PART3 + NWG_PAIR*256*4)    // [NB][256]       x4 stats partials
#define WS_SC     (WS_PART4 + NB*256*4)          // 10*128 f32: sc0,bi0,sc1,bi1,...,sc4,bi4

// --------------------- LDS layout for the pair kernels ---------------------
#define L_W2   0
#define L_H1   32768
#define L_AI   65536           // 128 f32
#define L_SCL  66048           // 6*128 f32 (scA,biA,scB,biB,scC,biC)
#define L_RED  69120           // 2 * 8*128 f32 (sum / sumsq per wave)
#define SMEM_PAIR0 77312
#define L_W3   77312
#define L_H2   110080
#define SMEM_PAIR1 142848
#define SMEM_OUT   73728       // W4bf(32K) + Arel(32K) + red(8K)

static __device__ inline v8f vzero8() {
  v8f z = {0.f,0.f,0.f,0.f,0.f,0.f,0.f,0.f};
  return z;
}

// WMMA 128x128x128 bf16 GEMM: C(m,n) = sum_k A[m][k] * W[n][k].
// A,W row-major bf16 in LDS, 8 waves, wave w owns rows [16w,16w+16).
// Fragments follow the CDNA5 ISA VGPR layouts (CDNA5 ISA):
//  A 16x32: lane<16 -> M=lane, K = {half*8..+7, 16+half*8..+7}  (two b128 loads)
//  B 32x16: lane<16 -> N=lane, K = {half*8..+7, 16+half*8..+7}  (two b128 loads)
static __device__ inline void wmma128(const bf16_t* A, const bf16_t* W, v8f acc[8],
                                      int wave, int laneM, int laneHalf) {
  const int mBase = wave * 16;
#pragma unroll
  for (int k0 = 0; k0 < HD; k0 += 32) {
    const bf16_t* pa = A + (mBase + laneM) * HD + k0 + laneHalf * 8;
    v8bf alo = *(const v8bf*)pa;
    v8bf ahi = *(const v8bf*)(pa + 16);
    v16bf af = __builtin_shufflevector(alo, ahi, 0,1,2,3,4,5,6,7,8,9,10,11,12,13,14,15);
#pragma unroll
    for (int nt = 0; nt < 8; ++nt) {
      const bf16_t* pb = W + (nt * 16 + laneM) * HD + k0 + laneHalf * 8;
      v8bf blo = *(const v8bf*)pb;
      v8bf bhi = *(const v8bf*)(pb + 16);
      v16bf bfr = __builtin_shufflevector(blo, bhi, 0,1,2,3,4,5,6,7,8,9,10,11,12,13,14,15);
      acc[nt] = __builtin_amdgcn_wmma_f32_16x16x32_bf16(
          false, af, false, bfr, (short)0, acc[nt], false, false);
    }
  }
}

// Per-channel sum / sumsq of a 128x128 accumulator tile set -> redS/redQ[8][128].
// C layout: element (r, lane) -> row mBase + 8*laneHalf + r, col nt*16 + laneM;
// lanes L and L^16 share the channel -> combine with shfl_xor(16).
static __device__ inline void stats_from_acc(const v8f acc[8], float* redS, float* redQ,
                                             int wave, int laneM, int lane) {
#pragma unroll
  for (int nt = 0; nt < 8; ++nt) {
    float s = 0.f, q = 0.f;
#pragma unroll
    for (int r = 0; r < 8; ++r) { float v = acc[nt][r]; s += v; q += v * v; }
    s += __shfl_xor(s, 16);
    q += __shfl_xor(q, 16);
    if (lane < 16) {
      redS[wave * 128 + nt * 16 + laneM] = s;
      redQ[wave * 128 + nt * 16 + laneM] = q;
    }
  }
}

// ------------------------- small helper kernels ----------------------------
__global__ void k_prep(const float* W2, const float* W3, const float* W4, char* ws) {
  bf16_t* w2 = (bf16_t*)(ws + WS_W2BF);
  bf16_t* w3 = (bf16_t*)(ws + WS_W3BF);
  bf16_t* w4 = (bf16_t*)(ws + WS_W4BF);
  for (int e = threadIdx.x; e < HD * HD; e += blockDim.x) {
    w2[e] = (bf16_t)W2[e]; w3[e] = (bf16_t)W3[e]; w4[e] = (bf16_t)W4[e];
  }
}

// a = pts@W1^T, s_pre = pts@W0^T (K=4, VALU), plus per-b partials for the
// analytic layer-1 BN variance (mean is exactly 0) and s-path stats.
__global__ __launch_bounds__(128) void k_point(const float* ctx, const float* W1,
                                               const float* W0, char* ws) {
  const int b = blockIdx.x, c = threadIdx.x;
  float w1r[NF], w0r[NF];
#pragma unroll
  for (int f = 0; f < NF; ++f) { w1r[f] = W1[c * NF + f]; w0r[f] = W0[c * NF + f]; }
  float S1 = 0.f, S2 = 0.f, Ss = 0.f, Sq = 0.f;
  float* a = (float*)(ws + WS_A);
  float* s = (float*)(ws + WS_S);
  for (int i = 0; i < NO; ++i) {
    const float* p = ctx + (size_t)b * NO * NF + i * NF;
    float p0 = p[0], p1 = p[1], p2 = p[2], p3 = p[3];
    float av = p0 * w1r[0] + p1 * w1r[1] + p2 * w1r[2] + p3 * w1r[3];
    float sv = p0 * w0r[0] + p1 * w0r[1] + p2 * w0r[2] + p3 * w0r[3];
    int r = b * NO + i;
    a[(size_t)r * HD + c] = av;
    s[(size_t)r * HD + c] = sv;
    S1 += av; S2 += av * av; Ss += sv; Sq += sv * sv;
  }
  ((float*)(ws + WS_VAR1P))[b * HD + c] = 2.f * NO * S2 - 2.f * S1 * S1;
  ((float*)(ws + WS_SSUMP))[b * HD + c] = Ss;
  ((float*)(ws + WS_SSQP))[b * HD + c]  = Sq;
}

__global__ __launch_bounds__(128) void k_stats1(const float* g1, const float* b1,
                                                const float* g0, const float* b0, char* ws) {
  const int c = threadIdx.x;
  const float* v1p = (const float*)(ws + WS_VAR1P);
  const float* ssp = (const float*)(ws + WS_SSUMP);
  const float* sqp = (const float*)(ws + WS_SSQP);
  float v1 = 0.f, ss = 0.f, sq = 0.f;
  for (int b = 0; b < NB; ++b) { v1 += v1p[b*HD+c]; ss += ssp[b*HD+c]; sq += sqp[b*HD+c]; }
  float* sc = (float*)(ws + WS_SC);
  float var1 = v1 / NPAIRF;                        // mean1 == 0 exactly
  sc[2*128 + c] = g1[c] * rsqrtf(var1 + EPSBN);    // sc1
  sc[3*128 + c] = b1[c];                           // bi1
  float ms = ss / (float)NROW;
  float vs = sq / (float)NROW - ms * ms;
  float s0 = g0[c] * rsqrtf(vs + EPSBN);
  sc[0*128 + c] = s0;                              // sc0
  sc[1*128 + c] = b0[c] - ms * s0;                 // bi0
}

__global__ __launch_bounds__(128) void k_statsN(const float* part, int nwg, float invN,
                                                const float* g, const float* beta,
                                                float* scOut, float* biOut) {
  const int c = threadIdx.x;
  float s = 0.f, q = 0.f;
  for (int w = 0; w < nwg; ++w) { s += part[w*256 + c]; q += part[w*256 + 128 + c]; }
  float m = s * invN;
  float v = q * invN - m * m;
  float sc = g[c] * rsqrtf(v + EPSBN);
  scOut[c] = sc;
  biOut[c] = beta[c] - m * sc;
}

// ------------------- fused all-pairs passes (heavy WMMA) -------------------
// MODE 0: h1 -> x2 = h1*W2^T,      accumulate x2 sum/sumsq     (stats for BN2)
// MODE 1: h1 -> h2 -> x3 = h2*W3^T, accumulate x3 sum/sumsq    (stats for BN3)
// MODE 2: h1 -> h2 -> h3, masked column-sum (j != i) -> v_rel
template <int MODE>
__global__ __launch_bounds__(256) void k_pair(char* ws) {
  extern __shared__ char smem[];
  bf16_t* W2l = (bf16_t*)(smem + L_W2);
  bf16_t* h1  = (bf16_t*)(smem + L_H1);
  float*  ai  = (float*) (smem + L_AI);
  float*  scA = (float*) (smem + L_SCL);
  float*  biA = scA + 128;
  float*  scB = scA + 256; float* biB = scA + 384;
  float*  scC = scA + 512; float* biC = scA + 640;
  float*  redS = (float*)(smem + L_RED);
  float*  redQ = redS + 1024;
  bf16_t* W3l = (bf16_t*)(smem + L_W3);
  bf16_t* h2  = (bf16_t*)(smem + L_H2);

  const int tid = threadIdx.x;
  const int wave = tid >> 5, lane = tid & 31, laneM = lane & 15, laneHalf = lane >> 4;
  const float* a = (const float*)(ws + WS_A);
  const float* scales = (const float*)(ws + WS_SC);
  float* vrel = (float*)(ws + WS_VREL);

  // stage weights and BN scale/bias
  {
    const unsigned short* src = (const unsigned short*)(ws + WS_W2BF);
    unsigned short* dst = (unsigned short*)W2l;
    for (int e = tid; e < HD * HD; e += 256) dst[e] = src[e];
    if constexpr (MODE > 0) {
      src = (const unsigned short*)(ws + WS_W3BF);
      dst = (unsigned short*)W3l;
      for (int e = tid; e < HD * HD; e += 256) dst[e] = src[e];
    }
    if (tid < 128) {
      scA[tid] = scales[2*128 + tid]; biA[tid] = scales[3*128 + tid];
      if constexpr (MODE > 0)  { scB[tid] = scales[4*128 + tid]; biB[tid] = scales[5*128 + tid]; }
      if constexpr (MODE == 2) { scC[tid] = scales[6*128 + tid]; biC[tid] = scales[7*128 + tid]; }
    }
  }
  float accSum = 0.f, accSq = 0.f;   // per-channel running stats (tid<128)
  __syncthreads();

  const int tilesPerWG = NROW / NWG_PAIR;
  for (int tt = 0; tt < tilesPerWG; ++tt) {
    const int t = blockIdx.x * tilesPerWG + tt;     // (b,i) tile
    const int b = t >> 7, irow = t & 127;

    if (tid < 128) ai[tid] = a[(size_t)t * HD + tid];
    __syncthreads();

    // h1[j][c] = relu((a_i[c]-a_j[c])*sc1[c]+bi1[c]) -> bf16 LDS (mean1==0)
    for (int e = tid; e < NO * HD; e += 256) {
      int j = e >> 7, c = e & 127;
      float x = ai[c] - a[((size_t)(b << 7) + j) * HD + c];
      h1[e] = (bf16_t)fmaxf(fmaf(x, scA[c], biA[c]), 0.f);
    }
    __syncthreads();

    v8f acc[8];
#pragma unroll
    for (int nt = 0; nt < 8; ++nt) acc[nt] = vzero8();
    wmma128(h1, W2l, acc, wave, laneM, laneHalf);   // x2

    if constexpr (MODE == 0) {
      stats_from_acc(acc, redS, redQ, wave, laneM, lane);
      __syncthreads();
      if (tid < 128) {
        float s = 0.f, q = 0.f;
        for (int w = 0; w < 8; ++w) { s += redS[w*128 + tid]; q += redQ[w*128 + tid]; }
        accSum += s; accSq += q;
      }
      __syncthreads();
    } else {
      // h2 = relu(x2*sc2+bi2) -> bf16 LDS
      const int mB = wave * 16 + laneHalf * 8;
#pragma unroll
      for (int nt = 0; nt < 8; ++nt) {
        int c = nt * 16 + laneM;
        float scv = scB[c], biv = biB[c];
#pragma unroll
        for (int r = 0; r < 8; ++r)
          h2[(mB + r) * HD + c] = (bf16_t)fmaxf(fmaf(acc[nt][r], scv, biv), 0.f);
      }
      __syncthreads();
#pragma unroll
      for (int nt = 0; nt < 8; ++nt) acc[nt] = vzero8();
      wmma128(h2, W3l, acc, wave, laneM, laneHalf); // x3

      if constexpr (MODE == 1) {
        stats_from_acc(acc, redS, redQ, wave, laneM, lane);
        __syncthreads();
        if (tid < 128) {
          float s = 0.f, q = 0.f;
          for (int w = 0; w < 8; ++w) { s += redS[w*128 + tid]; q += redQ[w*128 + tid]; }
          accSum += s; accSq += q;
        }
        __syncthreads();
      } else {
        // h3 = relu(x3*sc3+bi3); v_rel[t][c] = sum_{j != i} h3[j][c]
        const int mB = wave * 16 + laneHalf * 8;
        float cs[8];
#pragma unroll
        for (int nt = 0; nt < 8; ++nt) {
          int c = nt * 16 + laneM;
          float scv = scC[c], biv = biC[c], s = 0.f;
#pragma unroll
          for (int r = 0; r < 8; ++r) {
            float v = fmaxf(fmaf(acc[nt][r], scv, biv), 0.f);
            s += ((mB + r) == irow) ? 0.f : v;    // exclude self-pair
          }
          cs[nt] = s;
        }
#pragma unroll
        for (int nt = 0; nt < 8; ++nt) cs[nt] += __shfl_xor(cs[nt], 16);
        if (lane < 16) {
#pragma unroll
          for (int nt = 0; nt < 8; ++nt) redS[wave*128 + nt*16 + laneM] = cs[nt];
        }
        __syncthreads();
        if (tid < 128) {
          float s = 0.f;
          for (int w = 0; w < 8; ++w) s += redS[w*128 + tid];
          vrel[(size_t)t * HD + tid] = s;
        }
        __syncthreads();
      }
    }
  }
  if constexpr (MODE < 2) {
    if (tid < 128) {
      float* part = (float*)(ws + (MODE == 0 ? WS_PART2 : WS_PART3));
      part[blockIdx.x * 256 + tid]       = accSum;
      part[blockIdx.x * 256 + 128 + tid] = accSq;
    }
  }
}

// x4 = v_rel @ W4^T (WMMA) + x4 stats partials; x4 stored into WS_A (a dead).
__global__ __launch_bounds__(256) void k_out1(char* ws) {
  extern __shared__ char smem[];
  bf16_t* W4l = (bf16_t*)smem;
  bf16_t* Ar  = (bf16_t*)(smem + 32768);
  float* redS = (float*)(smem + 65536);
  float* redQ = redS + 1024;
  const int tid = threadIdx.x;
  const int wave = tid >> 5, lane = tid & 31, laneM = lane & 15, laneHalf = lane >> 4;

  {
    const unsigned short* src = (const unsigned short*)(ws + WS_W4BF);
    unsigned short* dst = (unsigned short*)W4l;
    for (int e = tid; e < HD * HD; e += 256) dst[e] = src[e];
  }
  const float* vrel = (const float*)(ws + WS_VREL);
  for (int e = tid; e < HD * HD; e += 256)
    Ar[e] = (bf16_t)vrel[(size_t)blockIdx.x * HD * HD + e];
  __syncthreads();

  v8f acc[8];
#pragma unroll
  for (int nt = 0; nt < 8; ++nt) acc[nt] = vzero8();
  wmma128(Ar, W4l, acc, wave, laneM, laneHalf);

  float* x4 = (float*)(ws + WS_A);
  const int mB = wave * 16 + laneHalf * 8;
#pragma unroll
  for (int nt = 0; nt < 8; ++nt)
#pragma unroll
    for (int r = 0; r < 8; ++r)
      x4[((size_t)blockIdx.x * HD + mB + r) * HD + nt * 16 + laneM] = acc[nt][r];

  stats_from_acc(acc, redS, redQ, wave, laneM, lane);
  __syncthreads();
  if (tid < 128) {
    float s = 0.f, q = 0.f;
    for (int w = 0; w < 8; ++w) { s += redS[w*128 + tid]; q += redQ[w*128 + tid]; }
    float* part = (float*)(ws + WS_PART4);
    part[blockIdx.x * 256 + tid]       = s;
    part[blockIdx.x * 256 + 128 + tid] = q;
  }
}

// y = relu(s_pre*sc0+bi0) + relu(x4*sc4+bi4)
__global__ __launch_bounds__(256) void k_out2(char* ws, float* y) {
  size_t idx = (size_t)blockIdx.x * 256 + threadIdx.x;
  int c = (int)(idx & 127);
  const float* sc = (const float*)(ws + WS_SC);
  float s  = ((const float*)(ws + WS_S))[idx];
  float x4 = ((const float*)(ws + WS_A))[idx];
  float vs = fmaxf(fmaf(s,  sc[0*128 + c], sc[1*128 + c]), 0.f);
  float vr = fmaxf(fmaf(x4, sc[8*128 + c], sc[9*128 + c]), 0.f);
  y[idx] = vs + vr;
}

extern "C" void kernel_launch(void* const* d_in, const int* in_sizes, int n_in,
                              void* d_out, int out_size, void* d_ws, size_t ws_size,
                              hipStream_t stream) {
  (void)in_sizes; (void)n_in; (void)out_size; (void)ws_size; // ws needs ~27 MiB
  const float* ctx = (const float*)d_in[0];
  const float* W0 = (const float*)d_in[1];
  const float* g0 = (const float*)d_in[2];
  const float* b0 = (const float*)d_in[3];
  const float* W1 = (const float*)d_in[4];
  const float* g1 = (const float*)d_in[5];
  const float* b1 = (const float*)d_in[6];
  const float* W2 = (const float*)d_in[7];
  const float* g2 = (const float*)d_in[8];
  const float* b2 = (const float*)d_in[9];
  const float* W3 = (const float*)d_in[10];
  const float* g3 = (const float*)d_in[11];
  const float* b3 = (const float*)d_in[12];
  const float* W4 = (const float*)d_in[13];
  const float* g4 = (const float*)d_in[14];
  const float* b4 = (const float*)d_in[15];
  char* ws = (char*)d_ws;
  float* scales = (float*)(ws + WS_SC);

  k_prep<<<1, 256, 0, stream>>>(W2, W3, W4, ws);
  k_point<<<NB, 128, 0, stream>>>(ctx, W1, W0, ws);
  k_stats1<<<1, 128, 0, stream>>>(g1, b1, g0, b0, ws);

  k_pair<0><<<NWG_PAIR, 256, SMEM_PAIR0, stream>>>(ws);
  k_statsN<<<1, 128, 0, stream>>>((const float*)(ws + WS_PART2), NWG_PAIR,
                                  1.f / NPAIRF, g2, b2, scales + 4*128, scales + 5*128);
  k_pair<1><<<NWG_PAIR, 256, SMEM_PAIR1, stream>>>(ws);
  k_statsN<<<1, 128, 0, stream>>>((const float*)(ws + WS_PART3), NWG_PAIR,
                                  1.f / NPAIRF, g3, b3, scales + 6*128, scales + 7*128);
  k_pair<2><<<NWG_PAIR, 256, SMEM_PAIR1, stream>>>(ws);

  k_out1<<<NB, 256, SMEM_OUT, stream>>>(ws);
  k_statsN<<<1, 128, 0, stream>>>((const float*)(ws + WS_PART4), NB,
                                  1.f / (float)NROW, g4, b4, scales + 8*128, scales + 9*128);
  k_out2<<<(NROW * HD) / 256, 256, 0, stream>>>(ws, (float*)d_out);
}
